// Seq2SeqTagging_31473520345570
// MI455X (gfx1250) — matrix operand order repaired and checked by reading, not compiled, not run on hardware
//
#include <hip/hip_runtime.h>
#include <hip/hip_bf16.h>
#include <math.h>

#define B_ 256
#define T_ 128
#define H_ 512
#define L_ 4
#define V_ 32000
#define NT_ 74
#define NTP 80     // padded logits row width
#define NTW 128    // padded Wtag rows (multiple of 64)
#define BSTR 40    // LDS B-tile column stride in halfs (conflict-free padding)
#define GSTR 65    // LDS gate-tile row stride in floats (conflict-free padding)

typedef __attribute__((ext_vector_type(16))) _Float16 v16h;
typedef __attribute__((ext_vector_type(8)))  _Float16 v8h;
typedef __attribute__((ext_vector_type(8)))  float    v8f;

union AFrag { v16h v; v8h h[2]; };

// ---------------------------------------------------------------------------
// WMMA GEMM:  C[m,n] = scale * sum_k A[m,k] * W[n,k]  (+ bias[n])
// A split in K: k <  K0 -> A0 (lda0); k >= K0 -> A1 (lda1)  (fused [x|h] gates)
// W is [N,K] row-major f16.  Requirements: M % 128 == 0, K % 32 == 0,
// W readable up to 64*ceil(N/64) rows (pad allocations).
// Block = 256 threads (8 waves) -> tile 128(M) x 64(N); wave tile 16x64.
// B tile staged through double-buffered LDS; A and B software-pipelined one
// K-slab ahead (sched_barrier pins prefetch loads; unconditional ds_store
// keeps the staging load from being sunk).  sched_group_barriers interleave
// DS reads with WMMAs so the matrix pipe never waits a full LDS round-trip.
//
// Fused LSTM mode (cellC != null): weights are gate-interleaved (row 4j+p =
// gate p of hidden j), N = 4H.  Epilogue stages biased gates into LDS, then
// performs the LSTM cell update in-kernel: c (f32, in/out), h -> cellH (f16),
// optional cellDec[b, tStep, :] (f32).
// ---------------------------------------------------------------------------
__global__ void __launch_bounds__(256)
gemm_wmma(
    const _Float16* __restrict__ A0, long lda0, long strideA0,
    const _Float16* __restrict__ A1, long lda1, int K0,
    const _Float16* __restrict__ W,  long ldw,  long strideW,
    const float* __restrict__ bias, float scale,
    float* __restrict__ C32, long ldc32,
    _Float16* __restrict__ C16, long ldc16,
    long strideC, int M, int N, int K, int transT,
    float* __restrict__ cellC, _Float16* __restrict__ cellH,
    float* __restrict__ cellDec, int tStep)
{
    __shared__ __align__(16) _Float16 Bs[2][64 * BSTR];
    __shared__ float Gs[128 * GSTR];

    const int tid  = threadIdx.x;
    const int lane = tid & 31;
    const int wave = tid >> 5;
    const int z    = blockIdx.z;
    const int rowBase = blockIdx.x * 128 + wave * 16;
    const int colBase = blockIdx.y * 64;

    A0 += (long)z * strideA0;
    W  += (long)z * strideW;

    const int lh  = lane >> 4;    // 0 = lanes 0-15, 1 = lanes 16-31
    const int l15 = lane & 15;
    const int r   = rowBase + l15;
    const int kb  = lh ? 16 : 0;  // A-fragment K sub-offset for this lane half
    const int kbB = lh ? 16 : 0;  // B-fragment K sub-offset for this lane half

    // lane A row pointers, pre-offset by kb (fragment = [p, p+16))
    const _Float16* Arow0 = A0 + (long)r * lda0 + kb;
    const _Float16* Arow1 = A1 ? (A1 + (long)r * lda1 + kb) : (const _Float16*)0;

    // cooperative B staging: thread -> (col = tid/4, k-chunk = (tid&3)*8)
    const int scol  = tid >> 2;         // 0..63
    const int skoff = (tid & 3) * 8;    // 0,8,16,24
    const _Float16* wrow = W + (long)(colBase + scol) * ldw + skoff;
    const int sidx = scol * BSTR + skoff;

    // prologue: stage K-slab 0 into LDS buffer 0; load A fragment for slab 0
    v8h breg = *(const v8h*)(wrow);
    *(v8h*)(&Bs[0][sidx]) = breg;
    AFrag aCur;
    aCur.h[0] = *(const v8h*)(Arow0);
    aCur.h[1] = *(const v8h*)(Arow0 + 8);
    __syncthreads();

    v8f acc[4] = {v8f{}, v8f{}, v8f{}, v8f{}};
    const int nIter = K >> 5;

    for (int it = 0; it < nIter; ++it) {
        const int cur  = it & 1;
        const bool more = (it + 1) < nIter;
        // prefetch next K-slab unconditionally (index clamped to a valid 0 on
        // the final iteration)
        const int kk1 = more ? ((it + 1) << 5) : 0;
        v8h bnext = *(const v8h*)(wrow + kk1);
        const _Float16* ArowN = (kk1 < K0) ? (Arow0 + kk1) : (Arow1 + (kk1 - K0));
        AFrag aNext;
        aNext.h[0] = *(const v8h*)(ArowN);
        aNext.h[1] = *(const v8h*)(ArowN + 8);
        // pin the prefetch loads above the compute so they stay in flight
        // across the wmmas instead of being sunk next to their uses
        __builtin_amdgcn_sched_barrier(0);

        // all four B fragments from LDS into distinct registers
        const _Float16* sb = &Bs[cur][l15 * BSTR + kbB];
        AFrag bf0, bf1, bf2, bf3;
        bf0.h[0] = *(const v8h*)(sb);
        bf0.h[1] = *(const v8h*)(sb + 8);
        bf1.h[0] = *(const v8h*)(sb + 16 * BSTR);
        bf1.h[1] = *(const v8h*)(sb + 16 * BSTR + 8);
        bf2.h[0] = *(const v8h*)(sb + 32 * BSTR);
        bf2.h[1] = *(const v8h*)(sb + 32 * BSTR + 8);
        bf3.h[0] = *(const v8h*)(sb + 48 * BSTR);
        bf3.h[1] = *(const v8h*)(sb + 48 * BSTR + 8);

        acc[0] = __builtin_amdgcn_wmma_f32_16x16x32_f16(
            false, aCur.v, false, bf0.v, (short)0, acc[0], false, false);
        acc[1] = __builtin_amdgcn_wmma_f32_16x16x32_f16(
            false, aCur.v, false, bf1.v, (short)0, acc[1], false, false);
        acc[2] = __builtin_amdgcn_wmma_f32_16x16x32_f16(
            false, aCur.v, false, bf2.v, (short)0, acc[2], false, false);
        acc[3] = __builtin_amdgcn_wmma_f32_16x16x32_f16(
            false, aCur.v, false, bf3.v, (short)0, acc[3], false, false);

        // unconditional: last iteration writes the never-again-read buffer
        // (keeps load+store in one scheduling region, kills the branch)
        *(v8h*)(&Bs[cur ^ 1][sidx]) = bnext;

        // prescribe interleave: (4 DS rd)(wmma)(2 DS rd)(wmma)(2 DS rd)(wmma)
        // (wmma)(DS wr) -> >=2 fragment sets live, partial dscnt waits
        __builtin_amdgcn_sched_group_barrier(0x100, 4, 0);
        __builtin_amdgcn_sched_group_barrier(0x008, 1, 0);
        __builtin_amdgcn_sched_group_barrier(0x100, 2, 0);
        __builtin_amdgcn_sched_group_barrier(0x008, 1, 0);
        __builtin_amdgcn_sched_group_barrier(0x100, 2, 0);
        __builtin_amdgcn_sched_group_barrier(0x008, 1, 0);
        __builtin_amdgcn_sched_group_barrier(0x008, 1, 0);
        __builtin_amdgcn_sched_group_barrier(0x200, 1, 0);

        __syncthreads();
        aCur = aNext;
    }

    if (cellC) {
        // ---- fused LSTM cell epilogue ----
        // stage biased gates into LDS; col = 4*j + p (gate-interleaved layout)
        #pragma unroll
        for (int ns = 0; ns < 4; ++ns) {
            const int lcol = ns * 16 + l15;
            const float bcol = bias[colBase + lcol];
            #pragma unroll
            for (int j = 0; j < 8; ++j) {
                const int lrow = wave * 16 + (lh ? j + 8 : j);
                Gs[lrow * GSTR + lcol] = acc[ns][j] + bcol;
            }
        }
        __syncthreads();
        const int row  = tid & 127;                 // batch row within tile
        const int rg   = blockIdx.x * 128 + row;    // global batch row
        const int half = tid >> 7;                  // 0 or 1
        #pragma unroll
        for (int u = 0; u < 8; ++u) {
            const int jj = half * 8 + u;            // 0..15
            const float* gq = &Gs[row * GSTR + jj * 4];
            float si = 1.0f / (1.0f + expf(-gq[0]));
            float sf = 1.0f / (1.0f + expf(-gq[1]));
            float gg = gq[2];
            float so = 1.0f / (1.0f + expf(-gq[3]));
            const int jg = (colBase >> 2) + jj;     // global hidden index
            const long idx = (long)rg * H_ + jg;
            float cn = sf * cellC[idx] + si * tanhf(gg);
            float hn = so * tanhf(cn);
            cellC[idx] = cn;
            cellH[idx] = (_Float16)hn;
            if (cellDec) cellDec[((long)rg * T_ + tStep) * H_ + jg] = hn;
        }
        return;
    }

    #pragma unroll
    for (int ns = 0; ns < 4; ++ns) {
        #pragma unroll
        for (int j = 0; j < 8; ++j) {
            const int row = rowBase + (lh ? j + 8 : j);
            const int col = colBase + ns * 16 + l15;
            if (col < N) {
                float val = acc[ns][j] * scale + (bias ? bias[col] : 0.0f);
                if (C32) C32[(long)z * strideC + (long)row * ldc32 + col] = val;
                if (C16) {
                    if (transT > 0) {
                        long bb = row / transT, tt = row % transT;
                        C16[(bb * (long)N + col) * transT + tt] = (_Float16)val;
                    } else {
                        C16[(long)z * strideC + (long)row * ldc16 + col] = (_Float16)val;
                    }
                }
            }
        }
    }
}

__global__ void embed_kernel(const int* __restrict__ ids,
                             const float* __restrict__ emb,
                             _Float16* __restrict__ x16, long n)
{
    for (long idx = blockIdx.x * (long)blockDim.x + threadIdx.x; idx < n;
         idx += (long)gridDim.x * blockDim.x) {
        long row = idx / H_; int j = (int)(idx - row * H_);
        x16[idx] = (_Float16)emb[(long)ids[row] * H_ + j];
    }
}

// concat [Wih | Whh] along K with gate interleave: out row 4j+p = old row p*H+j
__global__ void conv_wcat(const float* __restrict__ Wih, const float* __restrict__ Whh,
                          _Float16* __restrict__ out, long n)
{
    for (long idx = blockIdx.x * (long)blockDim.x + threadIdx.x; idx < n;
         idx += (long)gridDim.x * blockDim.x) {
        int k   = (int)(idx % (2 * H_));
        long nn = idx / (2 * H_);
        int n4  = (int)(nn % (4 * H_));
        long l  = nn / (4 * H_);
        int j = n4 >> 2, ppp = n4 & 3;
        long srcrow = l * 4 * H_ + (long)ppp * H_ + j;
        float v = (k < H_) ? Wih[srcrow * H_ + k] : Whh[srcrow * H_ + (k - H_)];
        out[idx] = (_Float16)v;
    }
}

// summed LSTM bias, gate-interleaved: out[l,4j+p] = bih[l,p*H+j] + bhh[l,p*H+j]
__global__ void conv_bias_perm(const float* __restrict__ a, const float* __restrict__ b,
                               float* __restrict__ out, long n)
{
    for (long idx = blockIdx.x * (long)blockDim.x + threadIdx.x; idx < n;
         idx += (long)gridDim.x * blockDim.x) {
        int n4 = (int)(idx % (4 * H_));
        long l = idx / (4 * H_);
        int j = n4 >> 2, ppp = n4 & 3;
        long src = l * 4 * H_ + (long)ppp * H_ + j;
        out[idx] = a[src] + b[src];
    }
}

__global__ void conv_f16(const float* __restrict__ in, _Float16* __restrict__ out, long n)
{
    for (long idx = blockIdx.x * (long)blockDim.x + threadIdx.x; idx < n;
         idx += (long)gridDim.x * blockDim.x) out[idx] = (_Float16)in[idx];
}

__global__ void conv_wtag(const float* __restrict__ w, _Float16* __restrict__ out, long n)
{
    for (long idx = blockIdx.x * (long)blockDim.x + threadIdx.x; idx < n;
         idx += (long)gridDim.x * blockDim.x) {
        long row = idx / H_;
        out[idx] = (row < NT_) ? (_Float16)w[idx] : (_Float16)0.0f;
    }
}

__global__ void conv_btag(const float* __restrict__ b, float* __restrict__ out)
{
    int i = threadIdx.x;
    if (i < NTW) out[i] = (i < NT_) ? b[i] : 0.0f;
}

// one wave per (b,t) row of H=512
__global__ void layernorm_kernel(const float* __restrict__ x,
                                 const float* __restrict__ g, const float* __restrict__ bb,
                                 _Float16* __restrict__ out)
{
    int row  = blockIdx.x * 4 + (threadIdx.x >> 5);
    int lane = threadIdx.x & 31;
    const float* xr = x + (long)row * H_;
    float v[16], s = 0.f, sq = 0.f;
    #pragma unroll
    for (int i = 0; i < 16; ++i) { v[i] = xr[lane + i * 32]; s += v[i]; sq += v[i] * v[i]; }
    #pragma unroll
    for (int m = 16; m >= 1; m >>= 1) { s += __shfl_xor(s, m, 32); sq += __shfl_xor(sq, m, 32); }
    float mean = s * (1.0f / H_);
    float var  = sq * (1.0f / H_) - mean * mean;
    float rstd = rsqrtf(var + 1e-5f);
    _Float16* orow = out + (long)row * H_;
    #pragma unroll
    for (int i = 0; i < 16; ++i) {
        int col = lane + i * 32;
        orow[col] = (_Float16)((v[i] - mean) * rstd * g[col] + bb[col]);
    }
}

// one wave per (b,q) row of T=128 scores
__global__ void attn_softmax(const float* __restrict__ s, _Float16* __restrict__ a)
{
    int row  = blockIdx.x * 4 + (threadIdx.x >> 5);
    int lane = threadIdx.x & 31;
    const float* sr = s + (long)row * T_;
    float v[4], mx = -3.0e38f;
    #pragma unroll
    for (int i = 0; i < 4; ++i) { v[i] = sr[lane + i * 32]; mx = fmaxf(mx, v[i]); }
    #pragma unroll
    for (int m = 16; m >= 1; m >>= 1) mx = fmaxf(mx, __shfl_xor(mx, m, 32));
    float sum = 0.f;
    #pragma unroll
    for (int i = 0; i < 4; ++i) { v[i] = expf(v[i] - mx); sum += v[i]; }
    #pragma unroll
    for (int m = 16; m >= 1; m >>= 1) sum += __shfl_xor(sum, m, 32);
    float inv = 1.0f / sum;
    _Float16* ar = a + (long)row * T_;
    #pragma unroll
    for (int i = 0; i < 4; ++i) ar[lane + i * 32] = (_Float16)(v[i] * inv);
}

// one wave per (b,t) row: prob (74), nll accumulation
__global__ void loss_kernel(const float* __restrict__ logits,
                            const int* __restrict__ tag_ids,
                            const float* __restrict__ tag_mask,
                            float* __restrict__ prob, float* __restrict__ acc)
{
    int row  = blockIdx.x * 4 + (threadIdx.x >> 5);
    int lane = threadIdx.x & 31;
    const float* lr = logits + (long)row * NTP;
    float madd = (1.0f - tag_mask[row]) * -1e32f;
    float v[3], mx = -3.0e38f;
    #pragma unroll
    for (int i = 0; i < 3; ++i) {
        int col = lane + i * 32;
        v[i] = (col < NT_) ? (lr[col] + madd) : -3.0e38f;
        mx = fmaxf(mx, v[i]);
    }
    #pragma unroll
    for (int m = 16; m >= 1; m >>= 1) mx = fmaxf(mx, __shfl_xor(mx, m, 32));
    float sum = 0.f;
    #pragma unroll
    for (int i = 0; i < 3; ++i) {
        int col = lane + i * 32;
        v[i] = (col < NT_) ? expf(v[i] - mx) : 0.0f;
        sum += v[i];
    }
    #pragma unroll
    for (int m = 16; m >= 1; m >>= 1) sum += __shfl_xor(sum, m, 32);
    float inv = 1.0f / sum;
    #pragma unroll
    for (int i = 0; i < 3; ++i) {
        int col = lane + i * 32;
        if (col < NT_) prob[(long)row * NT_ + col] = v[i] * inv;
    }
    if (lane == 0) {
        int tag = tag_ids[row];
        float xt = lr[tag] + madd;
        float nll = -(xt - mx - logf(sum));
        float valid = (tag != 0) ? 1.0f : 0.0f;
        atomicAdd(&acc[0], nll * valid);
        atomicAdd(&acc[1], valid);
    }
}

__global__ void finalize_loss(const float* __restrict__ acc, float* __restrict__ out)
{
    out[0] = acc[0] / fmaxf(acc[1], 1.0f);
}

// ---------------------------------------------------------------------------
extern "C" void kernel_launch(void* const* d_in, const int* in_sizes, int n_in,
                              void* d_out, int out_size, void* d_ws, size_t ws_size,
                              hipStream_t stream)
{
    const int*   input_ids = (const int*)  d_in[0];
    const int*   tag_ids   = (const int*)  d_in[1];
    const float* tag_mask  = (const float*)d_in[2];
    const float* emb       = (const float*)d_in[3];
    const float* enc_Wih   = (const float*)d_in[4];
    const float* enc_Whh   = (const float*)d_in[5];
    const float* enc_bih   = (const float*)d_in[6];
    const float* enc_bhh   = (const float*)d_in[7];
    const float* dec_Wih   = (const float*)d_in[8];
    const float* dec_Whh   = (const float*)d_in[9];
    const float* dec_bih   = (const float*)d_in[10];
    const float* dec_bhh   = (const float*)d_in[11];
    const float* ln_g      = (const float*)d_in[12];
    const float* ln_b      = (const float*)d_in[13];
    const float* Wq        = (const float*)d_in[14];
    const float* bq        = (const float*)d_in[15];
    const float* Wk        = (const float*)d_in[16];
    const float* bk        = (const float*)d_in[17];
    const float* Wv        = (const float*)d_in[18];
    const float* bv        = (const float*)d_in[19];
    const float* Wtag      = (const float*)d_in[20];
    const float* btag      = (const float*)d_in[21];

    // ---- workspace carve (all 256-B aligned) ----
    char* p = (char*)d_ws;
    auto carve = [&](size_t bytes) -> char* {
        char* r = p; p += (bytes + 255) & ~(size_t)255; return r;
    };
    const long LH4x2 = (long)L_ * 4 * H_ * 2 * H_;
    _Float16* wcat_enc = (_Float16*)carve(LH4x2 * 2);
    _Float16* wcat_dec = (_Float16*)carve(LH4x2 * 2);
    float*    bias_enc = (float*)   carve((long)L_ * 4 * H_ * 4);
    float*    bias_dec = (float*)   carve((long)L_ * 4 * H_ * 4);
    _Float16* wq16     = (_Float16*)carve((long)H_ * H_ * 2);
    _Float16* wk16     = (_Float16*)carve((long)H_ * H_ * 2);
    _Float16* wv16     = (_Float16*)carve((long)H_ * H_ * 2);
    _Float16* wtag16   = (_Float16*)carve((long)NTW * H_ * 2);
    float*    btagp    = (float*)   carve((long)NTW * 4);
    _Float16* x16      = (_Float16*)carve((long)B_ * T_ * H_ * 2);
    _Float16* h16a     = (_Float16*)carve((long)L_ * B_ * H_ * 2);
    _Float16* h16b     = (_Float16*)carve((long)L_ * B_ * H_ * 2);
    float*    cst      = (float*)   carve((long)L_ * B_ * H_ * 4);
    float*    dec_out  = (float*)   carve((long)B_ * T_ * H_ * 4);
    _Float16* norm16   = (_Float16*)carve((long)B_ * T_ * H_ * 2);
    _Float16* q16      = (_Float16*)carve((long)B_ * T_ * H_ * 2);
    _Float16* k16      = (_Float16*)carve((long)B_ * T_ * H_ * 2);
    _Float16* vT16     = (_Float16*)carve((long)B_ * H_ * T_ * 2);
    float*    scores   = (float*)   carve((long)B_ * T_ * T_ * 4);
    _Float16* attn16   = (_Float16*)carve((long)B_ * T_ * T_ * 2);
    _Float16* ao16     = (_Float16*)carve((long)B_ * T_ * H_ * 2);
    float*    logits   = (float*)   carve((long)B_ * T_ * NTP * 4);
    float*    acc      = (float*)   carve(2 * 4);

    // ---- per-call init (graph-capture safe) ----
    hipMemsetAsync(cst, 0, (long)L_ * B_ * H_ * 4, stream);
    hipMemsetAsync(h16a, 0, (long)L_ * B_ * H_ * 2, stream);
    hipMemsetAsync(acc, 0, 8, stream);

    // ---- weight conversion ----
    conv_wcat<<<2048, 256, 0, stream>>>(enc_Wih, enc_Whh, wcat_enc, LH4x2);
    conv_wcat<<<2048, 256, 0, stream>>>(dec_Wih, dec_Whh, wcat_dec, LH4x2);
    conv_bias_perm<<<32, 256, 0, stream>>>(enc_bih, enc_bhh, bias_enc, (long)L_ * 4 * H_);
    conv_bias_perm<<<32, 256, 0, stream>>>(dec_bih, dec_bhh, bias_dec, (long)L_ * 4 * H_);
    conv_f16 <<<512, 256, 0, stream>>>(Wq, wq16, (long)H_ * H_);
    conv_f16 <<<512, 256, 0, stream>>>(Wk, wk16, (long)H_ * H_);
    conv_f16 <<<512, 256, 0, stream>>>(Wv, wv16, (long)H_ * H_);
    conv_wtag<<<256, 256, 0, stream>>>(Wtag, wtag16, (long)NTW * H_);
    conv_btag<<<1, NTW, 0, stream>>>(btag, btagp);
    embed_kernel<<<4096, 256, 0, stream>>>(input_ids, emb, x16, (long)B_ * T_ * H_);

    auto gemm = [&](const _Float16* A0, long lda0, long sA0,
                    const _Float16* A1, long lda1, int K0,
                    const _Float16* W, long ldw, long sW,
                    const float* bias, float scale,
                    float* C32, long ldc32, _Float16* C16, long ldc16,
                    long sC, int M, int N, int K, int transT, int batch,
                    float* cC, _Float16* cH, float* cD, int tStep) {
        dim3 g(M / 128, (N + 63) / 64, batch);
        gemm_wmma<<<g, dim3(256), 0, stream>>>(A0, lda0, sA0, A1, lda1, K0,
                                               W, ldw, sW, bias, scale,
                                               C32, ldc32, C16, ldc16,
                                               sC, M, N, K, transT,
                                               cC, cH, cD, tStep);
    };

    const long BH = (long)B_ * H_;
    _Float16* hb[2] = { h16a, h16b };

    // ---- encoder: T steps x L layers (fused GEMM + cell) ----
    for (int t = 0; t < T_; ++t) {
        const int par = t & 1;
        for (int l = 0; l < L_; ++l) {
            const _Float16* A0 = (l == 0) ? (x16 + (long)t * H_)
                                          : (hb[par ^ 1] + (long)(l - 1) * BH);
            long lda0 = (l == 0) ? (long)T_ * H_ : (long)H_;
            gemm(A0, lda0, 0, hb[par] + (long)l * BH, H_, H_,
                 wcat_enc + (long)l * 4 * H_ * 2 * H_, 2 * H_, 0,
                 bias_enc + l * 4 * H_, 1.0f,
                 nullptr, 0, nullptr, 0, 0, B_, 4 * H_, 2 * H_, 0, 1,
                 cst + (long)l * BH, hb[par ^ 1] + (long)l * BH, nullptr, t);
        }
    }

    // ---- decoder: carry (h,c) continues in place; xin == h[3] always ----
    for (int t = 0; t < T_; ++t) {
        const int par = t & 1;
        for (int l = 0; l < L_; ++l) {
            const _Float16* A0 = (l == 0) ? (hb[par] + 3L * BH)
                                          : (hb[par ^ 1] + (long)(l - 1) * BH);
            gemm(A0, H_, 0, hb[par] + (long)l * BH, H_, H_,
                 wcat_dec + (long)l * 4 * H_ * 2 * H_, 2 * H_, 0,
                 bias_dec + l * 4 * H_, 1.0f,
                 nullptr, 0, nullptr, 0, 0, B_, 4 * H_, 2 * H_, 0, 1,
                 cst + (long)l * BH, hb[par ^ 1] + (long)l * BH,
                 (l == 3) ? dec_out : nullptr, t);
        }
    }

    // ---- layernorm ----
    layernorm_kernel<<<B_ * T_ / 4, 128, 0, stream>>>(dec_out, ln_g, ln_b, norm16);

    // ---- Q,K,V projections (M = B*T) ----
    const int M = B_ * T_;
    gemm(norm16, H_, 0, nullptr, 0, H_, wq16, H_, 0, bq, 1.0f,
         nullptr, 0, q16, H_, 0, M, H_, H_, 0, 1, nullptr, nullptr, nullptr, 0);
    gemm(norm16, H_, 0, nullptr, 0, H_, wk16, H_, 0, bk, 1.0f,
         nullptr, 0, k16, H_, 0, M, H_, H_, 0, 1, nullptr, nullptr, nullptr, 0);
    gemm(norm16, H_, 0, nullptr, 0, H_, wv16, H_, 0, bv, 1.0f,
         nullptr, 0, vT16, 0, 0, M, H_, H_, T_, 1, nullptr, nullptr, nullptr, 0);

    // ---- scores = q k^T / sqrt(H), batched over B ----
    gemm(q16, H_, (long)T_ * H_, nullptr, 0, H_, k16, H_, (long)T_ * H_,
         nullptr, 0.04419417382415922f,
         scores, T_, nullptr, 0, (long)T_ * T_, T_, T_, H_, 0, B_,
         nullptr, nullptr, nullptr, 0);

    attn_softmax<<<B_ * T_ / 4, 128, 0, stream>>>(scores, attn16);

    // ---- attn_out = attn @ v  (W = v^T as [H,T]) ----
    gemm(attn16, T_, (long)T_ * T_, nullptr, 0, T_, vT16, T_, (long)H_ * T_,
         nullptr, 1.0f,
         nullptr, 0, ao16, H_, (long)T_ * H_, T_, H_, T_, 0, B_,
         nullptr, nullptr, nullptr, 0);

    // ---- logits = attn_out @ Wtag^T + btag (N padded to 80) ----
    gemm(ao16, H_, 0, nullptr, 0, H_, wtag16, H_, 0, btagp, 1.0f,
         logits, NTP, nullptr, 0, 0, M, NTP, H_, 0, 1,
         nullptr, nullptr, nullptr, 0);

    // ---- softmax / prob / loss ----
    loss_kernel<<<B_ * T_ / 4, 128, 0, stream>>>(logits, tag_ids, tag_mask,
                                                 (float*)d_out, acc);
    finalize_loss<<<1, 1, 0, stream>>>(acc, (float*)d_out + (size_t)B_ * T_ * NT_);

    (void)in_sizes; (void)n_in; (void)out_size; (void)ws_size;
}
